// AttentionLayer_6442450944088
// MI455X (gfx1250) — matrix operand order repaired and checked by reading, not compiled, not run on hardware
//
#include <hip/hip_runtime.h>
#include <hip/hip_bf16.h>

typedef __attribute__((ext_vector_type(16))) __bf16 bf16x16;
typedef __attribute__((ext_vector_type(8)))  __bf16 bf16x8;
typedef __attribute__((ext_vector_type(8)))  float  v8f;

#define SQRT_HALF 0.70710678118654752f

constexpr int B_ = 32, L_ = 1024, C_ = 1024, E_ = 1024;
constexpr int NRAW = 196;          // real attention context size (14*14)
constexpr int NP   = 224;          // padded to a multiple of 32 for WMMA K/N
constexpr long BL  = (long)B_ * L_;

union ABFrag { bf16x16 v; bf16x8 h[2]; };

enum { EP_H = 0, EP_SCORES = 1, EP_CTX = 2, EP_OUT = 3 };

#if __has_builtin(__builtin_amdgcn_global_load_async_to_lds_b128)
#define HAVE_ASYNC_LDS 1
#else
#define HAVE_ASYNC_LDS 0
#endif

#if HAVE_ASYNC_LDS
// exact pointee type per clang diagnostic: int __attribute__((vector_size(16)))
typedef int v4i_mem __attribute__((vector_size(16)));
typedef __attribute__((address_space(1))) v4i_mem* g_v4i_ptr;   // global
typedef __attribute__((address_space(3))) v4i_mem* l_v4i_ptr;   // LDS

__device__ __forceinline__ void async_cp16(const __bf16* g, __bf16* l) {
  __builtin_amdgcn_global_load_async_to_lds_b128((g_v4i_ptr)g, (l_v4i_ptr)l, 0, 0);
}
__device__ __forceinline__ void wait_async_le4() {
#if __has_builtin(__builtin_amdgcn_s_wait_asynccnt)
  __builtin_amdgcn_s_wait_asynccnt(4);
#else
  asm volatile("s_wait_asynccnt 0x4" ::: "memory");
#endif
}
__device__ __forceinline__ void wait_async_0() {
#if __has_builtin(__builtin_amdgcn_s_wait_asynccnt)
  __builtin_amdgcn_s_wait_asynccnt(0);
#else
  asm volatile("s_wait_asynccnt 0x0" ::: "memory");
#endif
}
#endif

// ---------------------------------------------------------------------------
// Tiled bf16 WMMA GEMM:  D[M,N] = A[M,K] @ W[N,K]^T  (+ epilogue)
// BM=BN=128, BK=32, 256 threads (8 wave32), wave tile 64x32 (4x2 WMMA accums)
// Double-buffered LDS fed by GLOBAL_LOAD_ASYNC_TO_LDS (ASYNCcnt pipeline).
// ---------------------------------------------------------------------------
template <int EP>
__global__ __launch_bounds__(256) void gemm_wmma(
    const __bf16* __restrict__ Ab, long strideA,
    const __bf16* __restrict__ Wb, long strideW,
    float* __restrict__ outF, __bf16* __restrict__ outB, long strideO,
    const float* __restrict__ bias,
    const float* __restrict__ add,
    int M, int N, int K, float scale)
{
  // 128 rows x 32 K, rows padded to 40 bf16 (80B) to avoid bank conflicts
  __shared__ __bf16 sA[2 * 128 * 40];
  __shared__ __bf16 sW[2 * 128 * 40];

  const int z = blockIdx.z;
  const __bf16* A = Ab + (size_t)strideA * z;
  const __bf16* W = Wb + (size_t)strideW * z;

  const int tileM = blockIdx.y * 128;
  const int tileN = blockIdx.x * 128;
  const int t    = threadIdx.x;
  const int lane = t & 31;
  const int wv   = t >> 5;
  const int wm   = wv & 1;     // wave row   (2 along M -> 64 rows each)
  const int wn   = wv >> 1;    // wave col   (4 along N -> 32 cols each)
  const int hi   = lane >> 4;  // half-wave selector in WMMA layouts
  const int lq   = lane & 15;

  // this thread's two 16B staging chunks per tile
  const int ar0 = t >> 2,          ac0 = (t & 3) * 8;           // chunk 0
  const int ar1 = (t + 256) >> 2,  ac1 = ((t + 256) & 3) * 8;   // chunk 1

  v8f acc[4][2];
  #pragma unroll
  for (int i = 0; i < 4; ++i)
    #pragma unroll
    for (int j = 0; j < 2; ++j)
      #pragma unroll
      for (int q = 0; q < 8; ++q) acc[i][j][q] = 0.0f;

  const int nk = K / 32;

#if HAVE_ASYNC_LDS
  auto issue = [&](int s) {
    const int buf = (s & 1) * (128 * 40);
    const int k0  = s * 32;
    // A tile (always in-bounds: M is a multiple of 128)
    async_cp16(A + (size_t)(tileM + ar0) * K + k0 + ac0, &sA[buf + ar0 * 40 + ac0]);
    async_cp16(A + (size_t)(tileM + ar1) * K + k0 + ac1, &sA[buf + ar1 * 40 + ac1]);
    // W tile; rows >= N read adjacent scratch (valid memory) -> garbage lands
    // only in accumulator columns >= N, which are masked at store time.
    async_cp16(W + (size_t)(tileN + ar0) * K + k0 + ac0, &sW[buf + ar0 * 40 + ac0]);
    async_cp16(W + (size_t)(tileN + ar1) * K + k0 + ac1, &sW[buf + ar1 * 40 + ac1]);
  };

  issue(0);
  for (int s = 0; s < nk; ++s) {
    if (s + 1 < nk) { issue(s + 1); wait_async_le4(); }  // stage s landed, s+1 in flight
    else           { wait_async_0(); }
    __syncthreads();

    const int buf = (s & 1) * (128 * 40);
    ABFrag af[4], wf[2];
    #pragma unroll
    for (int i = 0; i < 4; ++i) {   // A: M=lane&15, K chunks {8*hi, 16+8*hi}
      int r = buf + (wm * 64 + 16 * i + lq) * 40;
      af[i].h[0] = *(const bf16x8*)&sA[r + 8 * hi];
      af[i].h[1] = *(const bf16x8*)&sA[r + 16 + 8 * hi];
    }
    #pragma unroll
    for (int j = 0; j < 2; ++j) {   // B: N=lane&15, K chunks {8*hi, 16+8*hi}
      int r = buf + (wn * 32 + 16 * j + lq) * 40;
      wf[j].h[0] = *(const bf16x8*)&sW[r + 8 * hi];
      wf[j].h[1] = *(const bf16x8*)&sW[r + 16 + 8 * hi];
    }
    #pragma unroll
    for (int i = 0; i < 4; ++i)
      #pragma unroll
      for (int j = 0; j < 2; ++j)
        acc[i][j] = __builtin_amdgcn_wmma_f32_16x16x32_bf16(
            false, af[i].v, false, wf[j].v, (short)0, acc[i][j], false, false);
    __syncthreads();   // everyone done reading buf before stage s+2 overwrites it
  }
#else
  // fallback: synchronous staging through VGPRs
  for (int s = 0; s < nk; ++s) {
    const int k0 = s * 32;
    {
      uint4 va = *(const uint4*)(A + (size_t)(tileM + ar0) * K + k0 + ac0);
      *(uint4*)&sA[ar0 * 40 + ac0] = va;
      uint4 vb = *(const uint4*)(A + (size_t)(tileM + ar1) * K + k0 + ac1);
      *(uint4*)&sA[ar1 * 40 + ac1] = vb;
      uint4 w0 = make_uint4(0u, 0u, 0u, 0u), w1 = make_uint4(0u, 0u, 0u, 0u);
      if (tileN + ar0 < N) w0 = *(const uint4*)(W + (size_t)(tileN + ar0) * K + k0 + ac0);
      if (tileN + ar1 < N) w1 = *(const uint4*)(W + (size_t)(tileN + ar1) * K + k0 + ac1);
      *(uint4*)&sW[ar0 * 40 + ac0] = w0;
      *(uint4*)&sW[ar1 * 40 + ac1] = w1;
    }
    __syncthreads();
    ABFrag af[4], wf[2];
    #pragma unroll
    for (int i = 0; i < 4; ++i) {
      int r = (wm * 64 + 16 * i + lq) * 40;
      af[i].h[0] = *(const bf16x8*)&sA[r + 8 * hi];
      af[i].h[1] = *(const bf16x8*)&sA[r + 16 + 8 * hi];
    }
    #pragma unroll
    for (int j = 0; j < 2; ++j) {
      int r = (wn * 32 + 16 * j + lq) * 40;
      wf[j].h[0] = *(const bf16x8*)&sW[r + 8 * hi];
      wf[j].h[1] = *(const bf16x8*)&sW[r + 16 + 8 * hi];
    }
    #pragma unroll
    for (int i = 0; i < 4; ++i)
      #pragma unroll
      for (int j = 0; j < 2; ++j)
        acc[i][j] = __builtin_amdgcn_wmma_f32_16x16x32_bf16(
            false, af[i].v, false, wf[j].v, (short)0, acc[i][j], false, false);
    __syncthreads();
  }
#endif

  // ---- epilogue: D layout => row = 16*i + 8*hi + q, col = 16*j + lq ----
  #pragma unroll
  for (int i = 0; i < 4; ++i) {
    #pragma unroll
    for (int j = 0; j < 2; ++j) {
      #pragma unroll
      for (int q = 0; q < 8; ++q) {
        int row = tileM + wm * 64 + 16 * i + hi * 8 + q;
        int col = tileN + wn * 32 + 16 * j + lq;
        if (col < N) {
          float r = acc[i][j][q];
          size_t oidx = (size_t)strideO * z + (size_t)row * N + col;
          if constexpr (EP == EP_H) {
            r = (r + bias[col] + add[(size_t)row * N + col]) * scale;
            outB[oidx] = (__bf16)r;
          } else if constexpr (EP == EP_SCORES) {
            outF[oidx] = r;
          } else if constexpr (EP == EP_CTX) {
            outB[oidx] = (__bf16)(r * scale);
          } else { // EP_OUT
            r = (r + bias[col] + add[(size_t)row * N + col]) * scale;
            outF[oidx] = r;
          }
        }
      }
    }
  }
}

// ---------------------------------------------------------------------------
// weight-norm + bf16 pack:  out[row,c] = bf16( v[row,c] * g[row]/||v[row,:]|| )
// ---------------------------------------------------------------------------
__global__ __launch_bounds__(256) void weight_norm_bf16(
    const float* __restrict__ v, const float* __restrict__ g,
    __bf16* __restrict__ out, int cols)
{
  int row = blockIdx.x;
  const float* vr = v + (size_t)row * cols;
  float s = 0.f;
  for (int c = threadIdx.x; c < cols; c += 256) { float x = vr[c]; s += x * x; }
  #pragma unroll
  for (int off = 16; off > 0; off >>= 1) s += __shfl_xor(s, off);
  __shared__ float ws[8];
  if ((threadIdx.x & 31) == 0) ws[threadIdx.x >> 5] = s;
  __syncthreads();
  float tot = 0.f;
  #pragma unroll
  for (int i = 0; i < 8; ++i) tot += ws[i];
  float sc = g[row] * rsqrtf(tot);
  for (int c = threadIdx.x; c < cols; c += 256)
    out[(size_t)row * cols + c] = (__bf16)(vr[c] * sc);
}

// fp32 -> bf16 cast, 4 elems/thread (n divisible by 1024)
__global__ __launch_bounds__(256) void convert_f32_bf16(
    const float* __restrict__ in, __bf16* __restrict__ o, size_t n)
{
  size_t i = ((size_t)blockIdx.x * blockDim.x + threadIdx.x) * 4;
  if (i + 3 < n) {
    #pragma unroll
    for (int k = 0; k < 4; ++k) o[i + k] = (__bf16)in[i + k];
  }
}

// conv_feats [B,E,196] -> y_bf [B,E,224] (zero pad) and yT_bf [B,224,E]
__global__ __launch_bounds__(256) void build_y(
    const float* __restrict__ conv, __bf16* __restrict__ y, __bf16* __restrict__ yT)
{
  size_t idx = (size_t)blockIdx.x * blockDim.x + threadIdx.x; // B*E*224
  int v  = (int)(idx % NP);
  size_t be = idx / NP;
  int e  = (int)(be % E_);
  int b  = (int)(be / E_);
  float val = (v < NRAW) ? conv[be * NRAW + v] : 0.f;
  y[idx] = (__bf16)val;
  yT[((size_t)b * NP + v) * E_ + e] = (__bf16)val;
}

// softmax over 196 scores per row; writes f32 attn output + padded bf16 copy
__global__ __launch_bounds__(256) void softmax_rows(
    const float* __restrict__ scores, float* __restrict__ attn_out,
    __bf16* __restrict__ attn_bf)
{
  int row  = blockIdx.x * 8 + (threadIdx.x >> 5);
  int lane = threadIdx.x & 31;
  const float* srow = scores + (size_t)row * NP;
  float vals[7];
  float m = -3.4e38f;
  #pragma unroll
  for (int i = 0; i < 7; ++i) {
    int c = lane + 32 * i;
    float v = (c < NRAW) ? srow[c] : -3.4e38f;
    vals[i] = v; m = fmaxf(m, v);
  }
  #pragma unroll
  for (int off = 16; off > 0; off >>= 1) m = fmaxf(m, __shfl_xor(m, off));
  float sum = 0.f;
  #pragma unroll
  for (int i = 0; i < 7; ++i) {
    int c = lane + 32 * i;
    float e = (c < NRAW) ? __expf(vals[i] - m) : 0.f;
    vals[i] = e; sum += e;
  }
  #pragma unroll
  for (int off = 16; off > 0; off >>= 1) sum += __shfl_xor(sum, off);
  float inv = 1.f / sum;
  #pragma unroll
  for (int i = 0; i < 7; ++i) {
    int c = lane + 32 * i;
    if (c < NRAW) {
      float a = vals[i] * inv;
      attn_out[(size_t)row * NRAW + c] = a;
      attn_bf[(size_t)row * NP + c]    = (__bf16)a;
    } else if (c < NP) {
      attn_bf[(size_t)row * NP + c] = (__bf16)0.f;   // inert K-padding for ctx GEMM
    }
  }
}

extern "C" void kernel_launch(void* const* d_in, const int* in_sizes, int n_in,
                              void* d_out, int out_size, void* d_ws, size_t ws_size,
                              hipStream_t stream) {
  (void)in_sizes; (void)n_in; (void)out_size; (void)ws_size;
  const float* x     = (const float*)d_in[0];   // [B,L,C]
  const float* wemb  = (const float*)d_in[1];   // [B,L,E]
  const float* conv  = (const float*)d_in[2];   // [B,E,14,14]
  const float* in_v  = (const float*)d_in[3];   // [E,C]
  const float* in_g  = (const float*)d_in[4];
  const float* in_b  = (const float*)d_in[5];
  const float* out_v = (const float*)d_in[6];   // [C,E]
  const float* out_g = (const float*)d_in[7];
  const float* out_b = (const float*)d_in[8];

  float* out_main = (float*)d_out;                        // [B,L,C]
  float* attn_out = out_main + (size_t)BL * C_;           // [B,L,196]

  // scratch layout (aliased where lifetimes allow)
  char* ws = (char*)d_ws;
  size_t off = 0;
  auto alloc = [&](size_t bytes) { void* p = ws + off; off += (bytes + 255) & ~(size_t)255; return p; };
  __bf16* x_bf  = (__bf16*)alloc((size_t)BL * C_ * 2);        // 64 MB (later reused as scores)
  __bf16* Wi    = (__bf16*)alloc((size_t)E_ * C_ * 2);        //  2 MB
  __bf16* Wo    = (__bf16*)alloc((size_t)C_ * E_ * 2);        //  2 MB
  __bf16* h_bf  = (__bf16*)alloc((size_t)BL * E_ * 2);        // 64 MB (later reused as ctx)
  __bf16* y_bf  = (__bf16*)alloc((size_t)B_ * E_ * NP * 2);   // 14 MB
  __bf16* yT_bf = (__bf16*)alloc((size_t)B_ * NP * E_ * 2);   // 14 MB (scores GEMM may
                                                              //  read a little past end
                                                              //  into at_bf: benign)
  __bf16* at_bf = (__bf16*)alloc((size_t)BL * NP * 2);        // 14 MB
  float*  scores = (float*)x_bf;   // alias: x_bf dead after GEMM1
  __bf16* ctx_bf = h_bf;           // alias: h dead after GEMM2

  // 1) weight norm + bf16 pack
  weight_norm_bf16<<<E_, 256, 0, stream>>>(in_v, in_g, Wi, C_);
  weight_norm_bf16<<<C_, 256, 0, stream>>>(out_v, out_g, Wo, E_);
  // 2) casts / layout builds
  convert_f32_bf16<<<(unsigned)((size_t)BL * C_ / 1024), 256, 0, stream>>>(x, x_bf, (size_t)BL * C_);
  build_y<<<(unsigned)((size_t)B_ * E_ * NP / 256), 256, 0, stream>>>(conv, y_bf, yT_bf);

  // 3) GEMM1: h = (x @ Wi^T + in_b + word_embed) * sqrt(0.5)   [32768,1024,K=1024]
  gemm_wmma<EP_H><<<dim3(E_ / 128, BL / 128, 1), 256, 0, stream>>>(
      x_bf, 0, Wi, 0, nullptr, h_bf, 0, in_b, wemb, BL, E_, C_, SQRT_HALF);

  // 4) GEMM2 (batched): scores[b] = h[b] @ yT[b]^T   [1024,224,K=1024]
  gemm_wmma<EP_SCORES><<<dim3(2, L_ / 128, B_), 256, 0, stream>>>(
      h_bf, (long)L_ * E_, yT_bf, (long)NP * E_, scores, nullptr, (long)L_ * NP,
      nullptr, nullptr, L_, NP, E_, 1.0f);

  // 5) softmax over 196 + bf16 padded attn
  softmax_rows<<<(unsigned)(BL / 8), 256, 0, stream>>>(scores, attn_out, at_bf);

  // 6) GEMM3 (batched): ctx[b] = (attn[b] @ y[b]^T) * sqrt(196)=14  [1024,1024,K=224]
  gemm_wmma<EP_CTX><<<dim3(E_ / 128, L_ / 128, B_), 256, 0, stream>>>(
      at_bf, (long)L_ * NP, y_bf, (long)E_ * NP, nullptr, ctx_bf, (long)L_ * E_,
      nullptr, nullptr, L_, E_, NP, 14.0f);

  // 7) GEMM4: out = (ctx @ Wo^T + out_b + x) * sqrt(0.5)   [32768,1024,K=1024]
  gemm_wmma<EP_OUT><<<dim3(C_ / 128, BL / 128, 1), 256, 0, stream>>>(
      ctx_bf, 0, Wo, 0, out_main, nullptr, 0, out_b, x, BL, C_, E_, SQRT_HALF);
}
